// QuantizedLinear4bit_74594991997151
// MI455X (gfx1250) — matrix operand order repaired and checked by reading, not compiled, not run on hardware
//
#include <hip/hip_runtime.h>

// ---- CDNA5 (gfx1250) wave32 WMMA types ----
typedef __attribute__((ext_vector_type(16))) __bf16         v16bf;
typedef __attribute__((ext_vector_type(8)))  float          v8f;
typedef __attribute__((ext_vector_type(16))) unsigned short v16u;
typedef __attribute__((ext_vector_type(8)))  unsigned short v8u;
typedef __attribute__((ext_vector_type(4)))  float          f4;
typedef __attribute__((ext_vector_type(4)))  int            i4;
typedef __attribute__((ext_vector_type(4)))  unsigned int   u4;
typedef __attribute__((ext_vector_type(2)))  unsigned int   u2;

#define BM 128
#define BN 128
#define BK 64
#define THREADS 256
#define AS_STRIDE 72    // ushorts per As row: BK + 8 pad (pitch 144B, 16B aligned)
#define BS_STRIDE 136   // ushorts per Bs row: BN + 8 pad (pitch 272B, 16B aligned)

// ---- gfx1250 async copy global->LDS (GLOBAL_LOAD_ASYNC_TO_LDS_B128) ----
#if defined(__AMDGCN__) && __has_builtin(__builtin_amdgcn_global_load_async_to_lds_b128) && __has_builtin(__builtin_amdgcn_s_wait_asynccnt)
#define USE_ASYNC_LDS 1
typedef __attribute__((address_space(1))) i4 gi4;   // global v4i
typedef __attribute__((address_space(3))) i4 li4;   // LDS v4i
__device__ __forceinline__ void async_cp16(const void* g, void* l) {
  __builtin_amdgcn_global_load_async_to_lds_b128(
      (gi4*)(unsigned long long)g,
      (li4*)(unsigned int)(unsigned long long)l, 0, 0);
}
#endif

// fp32 -> bf16, round-to-nearest-even (manual fallback)
__device__ __forceinline__ unsigned short f2bf(float f) {
  unsigned int u = __float_as_uint(f);
  u += 0x7FFFu + ((u >> 16) & 1u);
  return (unsigned short)(u >> 16);
}

// packed 2x fp32 -> bf16x2
__device__ __forceinline__ unsigned int pk_bf16(float a, float b) {
#if defined(__AMDGCN__) && __has_builtin(__builtin_amdgcn_cvt_pk_bf16_f32)
  typedef __attribute__((ext_vector_type(2))) __bf16 v2bf;
  v2bf r = __builtin_amdgcn_cvt_pk_bf16_f32(a, b);
  return __builtin_bit_cast(unsigned int, r);
#else
  return (unsigned int)f2bf(a) | ((unsigned int)f2bf(b) << 16);
#endif
}

__device__ __forceinline__ v16bf mkfrag(v8u lo, v8u hi) {
  v16u t;
#pragma unroll
  for (int i = 0; i < 8; ++i) { t[i] = lo[i]; t[i + 8] = hi[i]; }
  return __builtin_bit_cast(v16bf, t);
}

// ===================== pass 1: x fp32 -> bf16 (once per element) ==============
__global__ __launch_bounds__(THREADS)
void cvt_x_bf16_kernel(const float* __restrict__ x, unsigned short* __restrict__ xb) {
  size_t base = ((size_t)blockIdx.x * THREADS + threadIdx.x) * 8;
  f4 a = *(const f4*)(x + base);
  f4 b = *(const f4*)(x + base + 4);
  u4 o;
  o[0] = pk_bf16(a[0], a[1]);
  o[1] = pk_bf16(a[2], a[3]);
  o[2] = pk_bf16(b[0], b[1]);
  o[3] = pk_bf16(b[2], b[3]);
  *(u4*)(xb + base) = o;
}

// ======== pass 2: dequant 4-bit W -> bf16, transposed to [K][O] (once) ========
__global__ __launch_bounds__(THREADS)
void dequant_wT_kernel(const int* __restrict__ wq, const float* __restrict__ sc,
                       const int* __restrict__ zp, unsigned short* __restrict__ wT,
                       int O, int K, int G) {
  const int o     = blockIdx.x * THREADS + threadIdx.x;
  const int kBase = blockIdx.y * 64;
  const int g     = kBase >> 7;                 // group_size = 128
  float s = sc[(size_t)o * G + g];
  float c = -s * (float)zp[(size_t)o * G + g];  // w = q*s + c
  const int* wp = wq + (size_t)o * (K >> 1) + (kBase >> 1);
#pragma unroll
  for (int j = 0; j < 8; ++j) {
    i4 v = *(const i4*)(wp + 4 * j);
#pragma unroll
    for (int cc = 0; cc < 4; ++cc) {
      int b = v[cc] & 0xFF;
      int k = kBase + (4 * j + cc) * 2;
      unsigned int p = pk_bf16(s * (float)(b & 15) + c,   // low nibble first
                               s * (float)(b >> 4) + c);
      wT[(size_t)(k + 0) * O + o] = (unsigned short)p;     // coalesced across lanes
      wT[(size_t)(k + 1) * O + o] = (unsigned short)(p >> 16);
    }
  }
}

// ========== pass 3: pure bf16 WMMA GEMM, double-buffered async staging ========
__global__ __launch_bounds__(THREADS)
void q4_gemm_bf16_kernel(const unsigned short* __restrict__ xb,
                         const unsigned short* __restrict__ wT,
                         const float* __restrict__ bias,
                         float* __restrict__ out, int M, int N, int K) {
  __shared__ unsigned short As[2][BM * AS_STRIDE];  // x tile  [m][k], ping-pong
  __shared__ unsigned short Bs[2][BK * BS_STRIDE];  // W tile  [k][n], ping-pong

  const int tid    = threadIdx.x;
  const int lane   = tid & 31;
  const int wave   = tid >> 5;
  const int waveM  = wave & 3;    // 4 waves along M
  const int waveN  = wave >> 2;   // 2 waves along N
  const int mBlock = blockIdx.y * BM;
  const int nBlock = blockIdx.x * BN;

  v8f acc[2][4];
#pragma unroll
  for (int mt = 0; mt < 2; ++mt)
#pragma unroll
    for (int nt = 0; nt < 4; ++nt)
#pragma unroll
      for (int r = 0; r < 8; ++r) acc[mt][nt][r] = 0.0f;

  // stage one K-chunk into buffer `buf` (8 x 16B segments per thread)
  auto stage = [&](int buf, int kBase) {
#pragma unroll
    for (int i = 0; i < 4; ++i) {
      int idx = tid + THREADS * i;
      int row = idx >> 3, seg = idx & 7;
      const unsigned short* src = xb + (size_t)(mBlock + row) * K + kBase + seg * 8;
      unsigned short* dst = &As[buf][row * AS_STRIDE + seg * 8];
#ifdef USE_ASYNC_LDS
      async_cp16(src, dst);
#else
      *(v8u*)dst = *(const v8u*)src;
#endif
    }
#pragma unroll
    for (int i = 0; i < 4; ++i) {
      int idx = tid + THREADS * i;
      int row = idx >> 4, seg = idx & 15;
      const unsigned short* src = wT + (size_t)(kBase + row) * N + nBlock + seg * 8;
      unsigned short* dst = &Bs[buf][row * BS_STRIDE + seg * 8];
#ifdef USE_ASYNC_LDS
      async_cp16(src, dst);
#else
      *(v8u*)dst = *(const v8u*)src;
#endif
    }
  };

  const int KT = K / BK;
  stage(0, 0);  // prologue: start loading tile 0

  for (int kt = 0; kt < KT; ++kt) {
    const int cur = kt & 1;

#ifdef USE_ASYNC_LDS
    __builtin_amdgcn_s_wait_asynccnt(0);   // own loads of tile kt have landed
#endif
    __syncthreads();                        // everyone's tile-kt data visible

    // kick off tile kt+1 into the other buffer; overlaps with WMMA below
    if (kt + 1 < KT) stage(1 - cur, (kt + 1) * BK);

    const unsigned short* Ab = As[cur];
    const unsigned short* Bb = Bs[cur];

    // ---- compute: 2 WMMA K-steps of 32, 2x4 tiles per wave ----
    const int lh = lane >> 4;
    const int lm = lane & 15;
#pragma unroll
    for (int ks = 0; ks < 2; ++ks) {
      const int k0 = ks * 32;
      v16bf afrag[2], bfrag[4];
#pragma unroll
      for (int mt = 0; mt < 2; ++mt) {
        // A 16x32 bf16: lanes 0-15 = rows; VGPR0-3: K = lh*8+0..7 ; VGPR4-7: +16
        const unsigned short* ap =
            &Ab[(waveM * 32 + mt * 16 + lm) * AS_STRIDE + k0 + lh * 8];
        afrag[mt] = mkfrag(*(const v8u*)ap, *(const v8u*)(ap + 16));
      }
#pragma unroll
      for (int nt = 0; nt < 4; ++nt) {
        // B 32x16 bf16: lane = N column; K = lh*8+0..7, then +16 (matches A)
        const unsigned short* bp =
            &Bb[(k0 + lh * 8) * BS_STRIDE + waveN * 64 + nt * 16 + lm];
        v16u bt;
#pragma unroll
        for (int i = 0; i < 8; ++i) { bt[i] = bp[i * BS_STRIDE]; bt[i + 8] = bp[(i + 16) * BS_STRIDE]; }
        bfrag[nt] = __builtin_bit_cast(v16bf, bt);
      }
#pragma unroll
      for (int mt = 0; mt < 2; ++mt)
#pragma unroll
        for (int nt = 0; nt < 4; ++nt)
          acc[mt][nt] = __builtin_amdgcn_wmma_f32_16x16x32_bf16(
              false, afrag[mt], false, bfrag[nt], (short)0, acc[mt][nt],
              false, false);
    }
    // no trailing barrier: next iteration's wait+barrier protects buffer reuse
  }

  // ---- epilogue: bias + store (lane<16 -> M=r, lane>=16 -> M=r+8) ----
  const int lh = lane >> 4;
  const int lm = lane & 15;
#pragma unroll
  for (int nt = 0; nt < 4; ++nt) {
    int nGlob = nBlock + waveN * 64 + nt * 16 + lm;
    float bv = bias[nGlob];
#pragma unroll
    for (int mt = 0; mt < 2; ++mt) {
      int mBase = mBlock + waveM * 32 + mt * 16 + lh * 8;
#pragma unroll
      for (int r = 0; r < 8; ++r)
        out[(size_t)(mBase + r) * N + nGlob] = acc[mt][nt][r] + bv;
    }
  }
}

// ============ fallback: fused single-pass kernel (ws too small) ===============
__global__ __launch_bounds__(THREADS)
void q4_linear_fused_kernel(const float* __restrict__ x, const int* __restrict__ wq,
                            const float* __restrict__ sc, const int* __restrict__ zp,
                            const float* __restrict__ bias, float* __restrict__ out,
                            int M, int N, int K, int G) {
  __shared__ unsigned short As[BM * AS_STRIDE];
  __shared__ unsigned short Bs[BK * BS_STRIDE];

  const int tid    = threadIdx.x;
  const int lane   = tid & 31;
  const int wave   = tid >> 5;
  const int waveM  = wave & 3;
  const int waveN  = wave >> 2;
  const int mBlock = blockIdx.y * BM;
  const int nBlock = blockIdx.x * BN;

  v8f acc[2][4];
#pragma unroll
  for (int mt = 0; mt < 2; ++mt)
#pragma unroll
    for (int nt = 0; nt < 4; ++nt)
#pragma unroll
      for (int r = 0; r < 8; ++r) acc[mt][nt][r] = 0.0f;

  const int wN    = tid >> 1;
  const int wHalf = tid & 1;

  const int KT = K / BK;
  for (int kt = 0; kt < KT; ++kt) {
    const int kBase = kt * BK;
#pragma unroll
    for (int i = 0; i < 8; ++i) {
      int idx = tid + THREADS * i;
      int row = idx >> 4, c4 = idx & 15;
      f4 v = *(const f4*)(x + (size_t)(mBlock + row) * K + kBase + c4 * 4);
      u2 p;
      p[0] = pk_bf16(v[0], v[1]);
      p[1] = pk_bf16(v[2], v[3]);
      *(u2*)&As[row * AS_STRIDE + c4 * 4] = p;
    }
    {
      const int g = kBase >> 7;
      float s = sc[(size_t)(nBlock + wN) * G + g];
      float c = -s * (float)zp[(size_t)(nBlock + wN) * G + g];
      const int* wp = wq + (size_t)(nBlock + wN) * (K >> 1) + (kBase >> 1) + wHalf * 16;
#pragma unroll
      for (int j = 0; j < 4; ++j) {
        i4 v = *(const i4*)(wp + 4 * j);
#pragma unroll
        for (int cc = 0; cc < 4; ++cc) {
          int b  = v[cc] & 0xFF;
          int kl = wHalf * 32 + (4 * j + cc) * 2;
          unsigned int p = pk_bf16(s * (float)(b & 15) + c, s * (float)(b >> 4) + c);
          Bs[(kl + 0) * BS_STRIDE + wN] = (unsigned short)p;
          Bs[(kl + 1) * BS_STRIDE + wN] = (unsigned short)(p >> 16);
        }
      }
    }
    __syncthreads();
    const int lh = lane >> 4, lm = lane & 15;
#pragma unroll
    for (int ks = 0; ks < 2; ++ks) {
      const int k0 = ks * 32;
      v16bf afrag[2], bfrag[4];
#pragma unroll
      for (int mt = 0; mt < 2; ++mt) {
        const unsigned short* ap =
            &As[(waveM * 32 + mt * 16 + lm) * AS_STRIDE + k0 + lh * 8];
        afrag[mt] = mkfrag(*(const v8u*)ap, *(const v8u*)(ap + 16));
      }
#pragma unroll
      for (int nt = 0; nt < 4; ++nt) {
        const unsigned short* bp =
            &Bs[(k0 + lh * 8) * BS_STRIDE + waveN * 64 + nt * 16 + lm];
        v16u bt;
#pragma unroll
        for (int i = 0; i < 8; ++i) { bt[i] = bp[i * BS_STRIDE]; bt[i + 8] = bp[(i + 16) * BS_STRIDE]; }
        bfrag[nt] = __builtin_bit_cast(v16bf, bt);
      }
#pragma unroll
      for (int mt = 0; mt < 2; ++mt)
#pragma unroll
        for (int nt = 0; nt < 4; ++nt)
          acc[mt][nt] = __builtin_amdgcn_wmma_f32_16x16x32_bf16(
              false, afrag[mt], false, bfrag[nt], (short)0, acc[mt][nt],
              false, false);
    }
    __syncthreads();
  }
  const int lh = lane >> 4, lm = lane & 15;
#pragma unroll
  for (int nt = 0; nt < 4; ++nt) {
    int nGlob = nBlock + waveN * 64 + nt * 16 + lm;
    float bv = bias[nGlob];
#pragma unroll
    for (int mt = 0; mt < 2; ++mt) {
      int mBase = mBlock + waveM * 32 + mt * 16 + lh * 8;
#pragma unroll
      for (int r = 0; r < 8; ++r)
        out[(size_t)(mBase + r) * N + nGlob] = acc[mt][nt][r] + bv;
    }
  }
}

extern "C" void kernel_launch(void* const* d_in, const int* in_sizes, int n_in,
                              void* d_out, int out_size, void* d_ws, size_t ws_size,
                              hipStream_t stream) {
  (void)n_in; (void)out_size;
  const float* x    = (const float*)d_in[0];
  const int*   wqp  = (const int*)d_in[1];
  const float* scp  = (const float*)d_in[2];
  const int*   zpp  = (const int*)d_in[3];
  const float* bsp  = (const float*)d_in[4];
  float*       outp = (float*)d_out;

  const int K = 4096;                 // in_features
  const int G = 32;                   // groups (group_size = 128)
  const int O = in_sizes[2] / G;      // 4096 out_features
  const int M = in_sizes[0] / K;      // 16384 flattened rows

  const size_t xbBytes = (size_t)M * K * 2;   // 128 MB
  const size_t wtBytes = (size_t)K * O * 2;   //  32 MB
  dim3 grid(O / BN, M / BM);

  if (ws_size >= xbBytes + wtBytes) {
    unsigned short* xb = (unsigned short*)d_ws;
    unsigned short* wT = (unsigned short*)((char*)d_ws + xbBytes);
    cvt_x_bf16_kernel<<<(unsigned)((size_t)M * K / (THREADS * 8)), THREADS, 0, stream>>>(x, xb);
    dequant_wT_kernel<<<dim3(O / THREADS, K / 64), THREADS, 0, stream>>>(wqp, scp, zpp, wT, O, K, G);
    q4_gemm_bf16_kernel<<<grid, dim3(THREADS), 0, stream>>>(xb, wT, bsp, outp, M, O, K);
  } else {
    q4_linear_fused_kernel<<<grid, dim3(THREADS), 0, stream>>>(x, wqp, scp, zpp, bsp, outp, M, O, K, G);
  }
}
